// CWFace_11201274708641
// MI455X (gfx1250) — matrix operand order repaired and checked by reading, not compiled, not run on hardware
//
#include <hip/hip_runtime.h>
#include <hip/hip_bf16.h>

typedef __attribute__((ext_vector_type(16))) __bf16 v16bf;
typedef __attribute__((ext_vector_type(8)))  __bf16 v8bf;
typedef __attribute__((ext_vector_type(4)))  __bf16 v4bf;
typedef __attribute__((ext_vector_type(8)))  float  v8f;

#define BATCH 512
#define EMB   512
#define NCLS  70722
#define TM    128   // rows per workgroup (8 waves x 16)
#define TC    64    // cols per workgroup (4 wmma col-tiles per wave)
#define TK    32    // k-step (bf16 wmma K)
#define BSTR  40    // LDS k-stride for B tiles (bf16): 80B, 16B-aligned rows

// ---------------- Phase 0: split embeddings f32 -> bf16 hi/lo planes ----------------
__global__ __launch_bounds__(256) void cwface_split_a(const float* __restrict__ emb,
                                                      __bf16* __restrict__ ahi,
                                                      __bf16* __restrict__ alo) {
    int i = blockIdx.x * 256 + threadIdx.x;   // 512*512 elements
    float f = emb[i];
    __bf16 h = (__bf16)f;
    ahi[i] = h;
    alo[i] = (__bf16)(f - (float)h);
}

// ---------------- Phase 1: column inverse norms of kernel [E, C] ----------------
__global__ __launch_bounds__(256) void cwface_colnorm(const float* __restrict__ kern,
                                                      float* __restrict__ invn) {
    int c = blockIdx.x * 256 + threadIdx.x;
    if (c >= NCLS) return;
    float s = 0.f;
    #pragma unroll 8
    for (int e = 0; e < EMB; ++e) {
        float v = kern[(size_t)e * NCLS + c];
        s = fmaf(v, v, s);
    }
    invn[c] = rsqrtf(s);
}

// ---------------- Phase 2: per-row margin stats (B=512, one block) ----------------
__global__ __launch_bounds__(512) void cwface_stats(const float* __restrict__ norms,
                                                    const int* __restrict__ label,
                                                    float* __restrict__ g_ang,
                                                    float* __restrict__ g_add) {
    __shared__ float sn[BATCH];
    __shared__ int   lb[BATCH];
    int t = threadIdx.x;
    float v = norms[t];
    v = fminf(fmaxf(v, 1e-3f), 100.0f);   // safe_norms
    sn[t] = v;
    lb[t] = label[t];
    __syncthreads();
    float sum = 0.f, sumsq = 0.f, gsum = 0.f, gcnt = 0.f;
    int myl = lb[t];
    for (int i = 0; i < BATCH; ++i) {
        float s = sn[i];
        sum += s; sumsq += s * s;
        if (lb[i] == myl) { gsum += s; gcnt += 1.f; }
    }
    float mean = sum * (1.0f / BATCH);
    float var  = fmaxf((sumsq - (float)BATCH * mean * mean) / (float)(BATCH - 1), 0.f);
    float stdv = sqrtf(var);              // T_ALPHA==1 -> new_std = std
    float m_b  = gsum / gcnt;             // group mean; count>=1 guaranteed
    float ms   = (v - m_b) / (stdv + 1e-3f) * 0.333f;
    ms = fminf(fmaxf(ms, -1.f), 1.f);
    g_ang[t] = -0.4f * ms;                // -M * margin_scaler
    g_add[t] =  0.4f + 0.4f * ms;         //  M + M * margin_scaler
}

// ---------------- Phase 3: WMMA GEMM + fused margin epilogue ----------------
__global__ __launch_bounds__(256) void cwface_gemm(const __bf16* __restrict__ ahi,
                                                   const __bf16* __restrict__ alo,
                                                   const float*  __restrict__ kern,
                                                   const float*  __restrict__ invn,
                                                   const float*  __restrict__ g_ang,
                                                   const float*  __restrict__ g_add,
                                                   const int*    __restrict__ label,
                                                   float* __restrict__ out) {
    // B tile staged as bf16 hi/lo, TRANSPOSED to k-major: fragment gather = ds_load_b128
    __shared__ alignas(16) __bf16 Bh[TC][BSTR];   // 5120 B
    __shared__ alignas(16) __bf16 Bl[TC][BSTR];   // 5120 B

    const int tid  = threadIdx.x;
    const int w    = tid >> 5;          // wave id 0..7 -> 16-row stripe
    const int lane = tid & 31;
    const int colBase = blockIdx.x * TC;
    const int rowBase = blockIdx.y * TM;

    v8f acc[4] = {};                    // four 16x16 f32 accumulators per wave

    // A 16-bit fragment addressing (ISA layout): lane -> row, k-base
    const int arow_g = rowBase + w * 16 + (lane & 15);
    const int kbase  = (lane >> 4) * 8;
    const __bf16* apH = ahi + (size_t)arow_g * EMB;
    const __bf16* apL = alo + (size_t)arow_g * EMB;
    // B 16-bit fragment addressing: lane -> col, k-base
    const int bn    = lane & 15;
    const int khalf = (lane >> 4) * 8;

    // cooperative B loader: this thread owns (c, k4) chunks of 4 k-consecutive elems
    for (int k0 = 0; k0 < EMB; k0 += TK) {
        #pragma unroll
        for (int i = 0; i < 2; ++i) {
            int lin = tid + i * 256;
            int cc  = lin & 63;
            int k4  = (lin >> 6) * 4;           // 0,4,...,28
            int c   = colBase + cc;
            float f0, f1, f2, f3;
            if (c < NCLS) {
                const float* kp = kern + (size_t)(k0 + k4) * NCLS + c;
                f0 = kp[0];
                f1 = kp[(size_t)NCLS];
                f2 = kp[(size_t)NCLS * 2];
                f3 = kp[(size_t)NCLS * 3];
            } else {
                f0 = f1 = f2 = f3 = 0.f;
            }
            __bf16 h0 = (__bf16)f0, h1 = (__bf16)f1, h2 = (__bf16)f2, h3 = (__bf16)f3;
            v4bf hv = { h0, h1, h2, h3 };
            v4bf lv = { (__bf16)(f0 - (float)h0), (__bf16)(f1 - (float)h1),
                        (__bf16)(f2 - (float)h2), (__bf16)(f3 - (float)h3) };
            *(v4bf*)&Bh[cc][k4] = hv;           // ds_store_b64
            *(v4bf*)&Bl[cc][k4] = lv;           // ds_store_b64
        }
        __syncthreads();

        // A fragments straight from global (L2-resident 1MB): 2x b128 each
        const __bf16* pH = apH + k0 + kbase;
        const __bf16* pL = apL + k0 + kbase;
        v8bf ah0 = *(const v8bf*)(pH);
        v8bf ah1 = *(const v8bf*)(pH + 16);
        v8bf al0 = *(const v8bf*)(pL);
        v8bf al1 = *(const v8bf*)(pL + 16);
        v16bf ah = __builtin_shufflevector(ah0, ah1, 0,1,2,3,4,5,6,7,8,9,10,11,12,13,14,15);
        v16bf al = __builtin_shufflevector(al0, al1, 0,1,2,3,4,5,6,7,8,9,10,11,12,13,14,15);

        #pragma unroll
        for (int ct = 0; ct < 4; ++ct) {
            const __bf16* pbh = &Bh[ct * 16 + bn][khalf];
            const __bf16* pbl = &Bl[ct * 16 + bn][khalf];
            v8bf bh0 = *(const v8bf*)(pbh);     // ds_load_b128
            v8bf bh1 = *(const v8bf*)(pbh + 16);
            v8bf bl0 = *(const v8bf*)(pbl);
            v8bf bl1 = *(const v8bf*)(pbl + 16);
            v16bf bh = __builtin_shufflevector(bh0, bh1, 0,1,2,3,4,5,6,7,8,9,10,11,12,13,14,15);
            v16bf bl = __builtin_shufflevector(bl0, bl1, 0,1,2,3,4,5,6,7,8,9,10,11,12,13,14,15);
            // acc += ah*bh + ah*bl + al*bh  (bf16x3: ~2^-16 rel error vs f32)
            acc[ct] = __builtin_amdgcn_wmma_f32_16x16x32_bf16(false, ah, false, bh,
                                                              (short)0, acc[ct], false, false);
            acc[ct] = __builtin_amdgcn_wmma_f32_16x16x32_bf16(false, ah, false, bl,
                                                              (short)0, acc[ct], false, false);
            acc[ct] = __builtin_amdgcn_wmma_f32_16x16x32_bf16(false, al, false, bh,
                                                              (short)0, acc[ct], false, false);
        }
        __syncthreads();
    }

    // ---------------- epilogue: scale by inv_norm, clip, rare margin path ----------------
    const int rhalf = (lane >> 4) * 8;        // D layout: VGPR r -> row r (+8 for hi lanes)
    float ga[8], gd[8]; int lb[8];
    #pragma unroll
    for (int r = 0; r < 8; ++r) {
        int row = rowBase + w * 16 + rhalf + r;
        ga[r] = g_ang[row]; gd[r] = g_add[row]; lb[r] = label[row];
    }
    const float CLIP = 1.0f - 1e-3f;
    const float PI_M = 3.14159265358979f - 1e-3f;
    #pragma unroll
    for (int ct = 0; ct < 4; ++ct) {
        int col = colBase + ct * 16 + (lane & 15);
        if (col >= NCLS) continue;
        float inv = invn[col];
        #pragma unroll
        for (int r = 0; r < 8; ++r) {
            int row = rowBase + w * 16 + rhalf + r;
            float cosv = acc[ct][r] * inv;
            cosv = fminf(fmaxf(cosv, -CLIP), CLIP);
            float o;
            if (col == lb[r]) {   // rare: one col per row out of 70722
                float th = acosf(cosv) + ga[r];
                th = fminf(fmaxf(th, 1e-3f), PI_M);
                o = (__builtin_cosf(th) - gd[r]) * 64.0f;
            } else {
                o = cosv * 64.0f; // cos(clip(acos(x))) == x here (clip no-op)
            }
            out[(size_t)row * NCLS + col] = o;
        }
    }
}

extern "C" void kernel_launch(void* const* d_in, const int* in_sizes, int n_in,
                              void* d_out, int out_size, void* d_ws, size_t ws_size,
                              hipStream_t stream) {
    const float* emb   = (const float*)d_in[0];   // [512, 512]
    const float* norms = (const float*)d_in[1];   // [512, 1]
    const int*   label = (const int*)d_in[2];     // [512]
    const float* kern  = (const float*)d_in[3];   // [512, 70722]
    // d_in[4] batch_mean, d_in[5] batch_std: unused (T_ALPHA == 1)

    // workspace layout (contiguous slabs): A_hi | A_lo | invn | g_ang | g_add
    __bf16* ahi  = (__bf16*)d_ws;                          // 512*512 bf16 = 512KB
    __bf16* alo  = ahi + (size_t)BATCH * EMB;              // 512KB
    float*  invn = (float*)(alo + (size_t)BATCH * EMB);    // C floats
    float*  g_ang = invn + NCLS;
    float*  g_add = g_ang + BATCH;
    float*  out   = (float*)d_out;                         // [B, C]

    cwface_split_a<<<(BATCH * EMB) / 256, 256, 0, stream>>>(emb, ahi, alo);
    cwface_colnorm<<<(NCLS + 255) / 256, 256, 0, stream>>>(kern, invn);
    cwface_stats<<<1, BATCH, 0, stream>>>(norms, label, g_ang, g_add);

    dim3 grid((NCLS + TC - 1) / TC, BATCH / TM);  // 1106 x 4
    cwface_gemm<<<grid, 256, 0, stream>>>(ahi, alo, kern, invn, g_ang, g_add, label, out);
}
